// compressor_51634096833375
// MI455X (gfx1250) — matrix operand order repaired and checked by reading, not compiled, not run on hardware
//
#include <hip/hip_runtime.h>

// Problem constants (from reference setup_inputs)
#define B_   8
#define S_   4096
#define DIN  1024
#define DOUT 512
#define CH   16            // cumsum chunk = one WMMA M-tile
#define NCH  (S_ / CH)     // 256 chunks per batch
#define WELEM (DOUT * DIN) // 524288 elements in each weight matrix

typedef __attribute__((ext_vector_type(16))) __bf16 v16bf;
typedef __attribute__((ext_vector_type(8)))  float  v8f;

// A fragment (16-bit A, 16x32 MxK layout): this lane holds row (lane&15),
// K = kbase..kbase+7 and kbase+16..kbase+23 where kbase = 8*(lane>>4).
// Source is f32 row-major; converts via hw v_cvt_pk_bf16_f32.
__device__ __forceinline__ v16bf load_a_frag_f32(const float* __restrict__ rowptr, int kbase) {
    const float4* p0 = reinterpret_cast<const float4*>(rowptr + kbase);
    const float4* p1 = reinterpret_cast<const float4*>(rowptr + kbase + 16);
    float4 a = p0[0], b = p0[1], c = p1[0], d = p1[1];
    v16bf f;
    f[0]=(__bf16)a.x;  f[1]=(__bf16)a.y;  f[2]=(__bf16)a.z;  f[3]=(__bf16)a.w;
    f[4]=(__bf16)b.x;  f[5]=(__bf16)b.y;  f[6]=(__bf16)b.z;  f[7]=(__bf16)b.w;
    f[8]=(__bf16)c.x;  f[9]=(__bf16)c.y;  f[10]=(__bf16)c.z; f[11]=(__bf16)c.w;
    f[12]=(__bf16)d.x; f[13]=(__bf16)d.y; f[14]=(__bf16)d.z; f[15]=(__bf16)d.w;
    return f;
}

// B fragment (16-bit B, 32x16 KxN): lane holds column (lane&15),
// K = kbase..kbase+7 and kbase+16..kbase+23 where kbase = 8*(lane>>4) (two 16B chunks).
__device__ __forceinline__ v16bf load_b_frag_bf16(const __bf16* __restrict__ p) {
    typedef __attribute__((ext_vector_type(8))) __bf16 v8bf;
    const v8bf lo = *reinterpret_cast<const v8bf*>(p);
    const v8bf hi = *reinterpret_cast<const v8bf*>(p + 16);
    return __builtin_shufflevector(lo, hi, 0, 1, 2, 3, 4, 5, 6, 7, 8, 9, 10, 11, 12, 13, 14, 15);
}

#define WMMA_BF16(A, Bv, C) \
    __builtin_amdgcn_wmma_f32_16x16x32_bf16(false, (A), false, (Bv), (short)0, (C), false, false)

// ---------------- Kernel 0: one-time weight conversion f32 -> bf16 -------------------
__global__ __launch_bounds__(256)
void cvt_w_kernel(const float* __restrict__ We, const float* __restrict__ Wd,
                  __bf16* __restrict__ we16, __bf16* __restrict__ wd16) {
    const int i = blockIdx.x * 256 + threadIdx.x;   // grid covers WELEM
    we16[i] = (__bf16)We[i];
    wd16[i] = (__bf16)Wd[i];
}

// ---------------- Kernel 1: encoder GEMM + bias + relu + mask + chunk partials --------
// grid (NCH, B), block 256 (8 waves). Wave w computes o-tiles [4w, 4w+4) for this 16-row chunk.
// k-loop unrolled x2 with ping-pong (double-buffered) A and B fragments: fragments for
// step k+32 are loaded while the WMMAs of step k execute, so no full-wait before WMMA.
__global__ __launch_bounds__(256)
void enc_kernel(const float* __restrict__ x, const float* __restrict__ mask,
                const __bf16* __restrict__ we16, const float* __restrict__ b_enc,
                float* __restrict__ h, float* __restrict__ partial) {
    const int b = blockIdx.y, chunk = blockIdx.x;
    const int s0 = chunk * CH;
    const int wave = threadIdx.x >> 5, lane = threadIdx.x & 31;
    const int half = lane >> 4, ln = lane & 15;

    const float*  xrow  = x + ((size_t)b * S_ + s0 + ln) * DIN;
    const __bf16* wbase = we16 + (size_t)(wave * 64 + ln) * DIN + half * 8;

    v8f acc[4] = {};
    v16bf af0 = load_a_frag_f32(xrow, half * 8);
    v16bf bf0[4], bf1[4];
#pragma unroll
    for (int t = 0; t < 4; ++t)
        bf0[t] = load_b_frag_bf16(wbase + (size_t)t * 16 * DIN);

    for (int k0 = 0; k0 < DIN; k0 += 64) {
        // ---- stage A: prefetch k0+32 (always in range: k0 <= DIN-64), compute k0 ----
        v16bf af1 = load_a_frag_f32(xrow, k0 + 32 + half * 8);
#pragma unroll
        for (int t = 0; t < 4; ++t)
            bf1[t] = load_b_frag_bf16(wbase + (size_t)t * 16 * DIN + k0 + 32);
#pragma unroll
        for (int t = 0; t < 4; ++t)
            acc[t] = WMMA_BF16(af0, bf0[t], acc[t]);

        // ---- stage B: prefetch k0+64 (wraps to 0 on last iter, dead), compute k0+32 --
        const int kn = (k0 + 64 < DIN) ? (k0 + 64) : 0;
        af0 = load_a_frag_f32(xrow, kn + half * 8);
#pragma unroll
        for (int t = 0; t < 4; ++t)
            bf0[t] = load_b_frag_bf16(wbase + (size_t)t * 16 * DIN + kn);
#pragma unroll
        for (int t = 0; t < 4; ++t)
            acc[t] = WMMA_BF16(af1, bf1[t], acc[t]);
    }

    // Epilogue. C/D layout: element r of lane -> row m = r + 8*(lane>>4), col n = lane&15.
    float maskv[8];
#pragma unroll
    for (int r = 0; r < 8; ++r)
        maskv[r] = mask[(size_t)b * S_ + s0 + half * 8 + r];

#pragma unroll
    for (int t = 0; t < 4; ++t) {
        const int col = (wave * 4 + t) * 16 + ln;
        const float be = b_enc[col];
        float csum = 0.f;
#pragma unroll
        for (int r = 0; r < 8; ++r) {
            const int s = s0 + half * 8 + r;
            float val = acc[t][r] + be;
            val = val > 0.f ? val : 0.f;
            val *= maskv[r];
            h[((size_t)b * S_ + s) * DOUT + col] = val;
            csum += val;
        }
        const float other = __shfl_xor(csum, 16, 32); // combine the two row-halves
        if (half == 0)
            partial[((size_t)b * NCH + chunk) * DOUT + col] = csum + other;
    }
}

// ---------------- Kernel 2: exclusive scan of chunk partials (in place) + v output ----
// One thread per (b, col): 8*512 = 4096 threads.
__global__ __launch_bounds__(256)
void scan_kernel(float* __restrict__ partial, float* __restrict__ v_out) {
    const int tid = blockIdx.x * 256 + threadIdx.x;
    const int b = tid / DOUT, col = tid % DOUT;
    float run = 0.f;
    for (int c = 0; c < NCH; ++c) {
        const size_t idx = ((size_t)b * NCH + c) * DOUT + col;
        const float p = partial[idx];
        partial[idx] = run;      // exclusive prefix
        run += p;
    }
    v_out[(size_t)b * DOUT + col] = run;   // v = total column sum = v_cum[:, -1, :]
}

// ---------------- Kernel 3: apply cumsum in place: h -> v_cum ------------------------
// grid (NCH, B), block 256; each thread owns 2 columns (coalesced across threads).
__global__ __launch_bounds__(256)
void cumsum_kernel(float* __restrict__ h, const float* __restrict__ partial) {
    const int b = blockIdx.y, chunk = blockIdx.x, s0 = chunk * CH;
#pragma unroll
    for (int cc = 0; cc < 2; ++cc) {
        const int col = threadIdx.x + 256 * cc;
        float acc = partial[((size_t)b * NCH + chunk) * DOUT + col];
        size_t base = ((size_t)b * S_ + s0) * DOUT + col;
#pragma unroll
        for (int r = 0; r < CH; ++r) {
            acc += h[base];
            h[base] = acc;
            base += DOUT;
        }
    }
}

// ---------------- Kernel 4: decoder GEMM + bias --------------------------------------
// grid (NCH, 2, B), block 256 (8 waves). Wave w handles d-tiles [dgrp*32 + 4w, +4).
// Same x2-unrolled double-buffered pipeline as the encoder.
__global__ __launch_bounds__(256)
void dec_kernel(const float* __restrict__ vcum, const __bf16* __restrict__ wd16,
                const float* __restrict__ b_dec, float* __restrict__ out) {
    const int b = blockIdx.z, chunk = blockIdx.x, dgrp = blockIdx.y;
    const int s0 = chunk * CH;
    const int wave = threadIdx.x >> 5, lane = threadIdx.x & 31;
    const int half = lane >> 4, ln = lane & 15;

    const float*  arow  = vcum + ((size_t)b * S_ + s0 + ln) * DOUT;
    const __bf16* wbase = wd16 + (size_t)((dgrp * 32 + wave * 4) * 16 + ln) * DOUT + half * 8;

    v8f acc[4] = {};
    v16bf af0 = load_a_frag_f32(arow, half * 8);
    v16bf bf0[4], bf1[4];
#pragma unroll
    for (int t = 0; t < 4; ++t)
        bf0[t] = load_b_frag_bf16(wbase + (size_t)t * 16 * DOUT);

    for (int k0 = 0; k0 < DOUT; k0 += 64) {
        v16bf af1 = load_a_frag_f32(arow, k0 + 32 + half * 8);
#pragma unroll
        for (int t = 0; t < 4; ++t)
            bf1[t] = load_b_frag_bf16(wbase + (size_t)t * 16 * DOUT + k0 + 32);
#pragma unroll
        for (int t = 0; t < 4; ++t)
            acc[t] = WMMA_BF16(af0, bf0[t], acc[t]);

        const int kn = (k0 + 64 < DOUT) ? (k0 + 64) : 0;
        af0 = load_a_frag_f32(arow, kn + half * 8);
#pragma unroll
        for (int t = 0; t < 4; ++t)
            bf0[t] = load_b_frag_bf16(wbase + (size_t)t * 16 * DOUT + kn);
#pragma unroll
        for (int t = 0; t < 4; ++t)
            acc[t] = WMMA_BF16(af1, bf1[t], acc[t]);
    }

#pragma unroll
    for (int t = 0; t < 4; ++t) {
        const int d = (dgrp * 32 + wave * 4 + t) * 16 + ln;
        const float bd = b_dec[d];
#pragma unroll
        for (int r = 0; r < 8; ++r) {
            const int s = s0 + half * 8 + r;
            out[((size_t)b * S_ + s) * DIN + d] = acc[t][r] + bd;
        }
    }
}

extern "C" void kernel_launch(void* const* d_in, const int* in_sizes, int n_in,
                              void* d_out, int out_size, void* d_ws, size_t ws_size,
                              hipStream_t stream) {
    const float* x     = (const float*)d_in[0];
    const float* mask  = (const float*)d_in[1];
    const float* W_enc = (const float*)d_in[2];
    const float* b_enc = (const float*)d_in[3];
    const float* W_dec = (const float*)d_in[4];
    const float* b_dec = (const float*)d_in[5];

    float* out     = (float*)d_out;
    float* v_out   = out;                        // (8, 512) first in tuple
    float* out_enc = out + (size_t)B_ * DOUT;    // (8, 4096, 1024)

    // Workspace layout (bytes):
    //   h/v_cum : B*S*DOUT * 4            = 64 MiB
    //   partial : B*NCH*DOUT * 4          =  4 MiB
    //   we16    : WELEM * 2               =  1 MiB
    //   wd16    : WELEM * 2               =  1 MiB
    float*  h       = (float*)d_ws;
    float*  partial = h + (size_t)B_ * S_ * DOUT;
    __bf16* we16    = (__bf16*)(partial + (size_t)B_ * NCH * DOUT);
    __bf16* wd16    = we16 + (size_t)WELEM;

    const dim3 blk(256);
    cvt_w_kernel<<<dim3(WELEM / 256), blk, 0, stream>>>(W_enc, W_dec, we16, wd16);
    enc_kernel<<<dim3(NCH, B_), blk, 0, stream>>>(x, mask, we16, b_enc, h, partial);
    scan_kernel<<<dim3((B_ * DOUT) / 256), blk, 0, stream>>>(partial, v_out);
    cumsum_kernel<<<dim3(NCH, B_), blk, 0, stream>>>(h, partial);
    dec_kernel<<<dim3(NCH, 2, B_), blk, 0, stream>>>(h, wd16, b_dec, out_enc);
}
